// RSSMCore_49460843381508
// MI455X (gfx1250) — compile-verified, matrix operand order checked by reading, not run on hardware
//
#include <hip/hip_runtime.h>

// ---------------------------------------------------------------------------
// RSSM core for gfx1250 (MI455X). All matmuls via v_wmma_f32_16x16x32_f16.
// Weights converted once per launch to f16 "fragment-packed" layout:
//   B (weights):  [ntile][kslab][lane(32)][e(16)]  halves
//   A (acts):     [kslab][mtile][lane(32)][e(16)]  halves
// where the (lane,e)->(row/col, k) map follows CDNA5 ISA 7.12.2 for 16-bit
// 16x32 A fragments:  k = ((e>>3)<<4) | ((lane>>4)<<3) | (e&7),  m/n = lane&15.
// C fragment (f32, 8 VGPRs): row = r + 8*(lane>>4), col = lane&15.
// ---------------------------------------------------------------------------

typedef __attribute__((ext_vector_type(16))) _Float16 v16h;
typedef __attribute__((ext_vector_type(8)))  float    v8f;

#define DEV __device__ __forceinline__

DEV v8f wmma32(v16h a, v16h b, v8f c) {
  return __builtin_amdgcn_wmma_f32_16x16x32_f16(
      /*neg_a=*/false, a, /*neg_b=*/false, b,
      /*c_mod=*/(short)0, c, /*reuse_a=*/false, /*reuse_b=*/false);
}

DEV int frag_k_map(int lane, int e) {
  return ((e >> 3) << 4) | (((lane >> 4) & 1) << 3) | (e & 7);
}

// Scatter one fp32 value into a packed f16 A-fragment buffer.
DEV void packA_store(_Float16* dst, size_t slabStride, int mt, int rr,
                     int kLocal, float v) {
  int ks = kLocal >> 5;
  int k  = kLocal & 31;
  int e  = ((k >> 4) << 3) | (k & 7);
  int ln = (((k >> 3) & 1) << 4) | rr;
  dst[(size_t)ks * slabStride + (size_t)mt * 512 + ln * 16 + e] = (_Float16)v;
}

DEV float sigmoid_f(float x) { return 1.f / (1.f + expf(-x)); }

// ---------------------------------------------------------------------------
// Weight pack: W fp32 [K,N] row-major -> packed f16 [NT][KS][32][16]
// ---------------------------------------------------------------------------
__global__ void pack_w_kernel(const float* __restrict__ W, _Float16* __restrict__ dst,
                              int KS, int Kreal, int N) {
  size_t total = (size_t)(N / 16) * KS * 512;
  for (size_t i = (size_t)blockIdx.x * blockDim.x + threadIdx.x; i < total;
       i += (size_t)gridDim.x * blockDim.x) {
    int e    = (int)(i & 15);
    int lane = (int)((i >> 4) & 31);
    size_t t = i >> 9;
    int ks = (int)(t % KS);
    int nt = (int)(t / KS);
    int n  = nt * 16 + (lane & 15);
    int k  = ks * 32 + frag_k_map(lane, e);
    float v = (k < Kreal) ? W[(size_t)k * N + n] : 0.f;
    dst[i] = (_Float16)v;
  }
}

// ---------------------------------------------------------------------------
// Activation pack: X fp32 [rows, ldx] -> packed f16 [KS][MT][32][16]
// ---------------------------------------------------------------------------
__global__ void pack_a_kernel(const float* __restrict__ X, int ldx,
                              _Float16* __restrict__ dst,
                              int MT, int KS, int Kreal) {
  size_t total = (size_t)KS * MT * 512;
  for (size_t i = (size_t)blockIdx.x * blockDim.x + threadIdx.x; i < total;
       i += (size_t)gridDim.x * blockDim.x) {
    int e    = (int)(i & 15);
    int lane = (int)((i >> 4) & 31);
    size_t t = i >> 9;
    int mtl = (int)(t % MT);
    int ks  = (int)(t / MT);
    int m = mtl * 16 + (lane & 15);
    int k = ks * 32 + frag_k_map(lane, e);
    float v = (k < Kreal) ? X[(size_t)m * ldx + k] : 0.f;
    dst[i] = (_Float16)v;
  }
}

// ---------------------------------------------------------------------------
// Fused GEMM (+optional second GEMM) + bias + LayerNorm + ELU.
// Grid: one block per 16-row tile; 512 threads = 16 waves, each wave owns 4
// column tiles of N=1024. Writes packed f16 A-fragments for the next stage.
// ---------------------------------------------------------------------------
__global__ __launch_bounds__(512)
void gemm_ln_kernel(const _Float16* __restrict__ A1, size_t sA1, int k1,
                    const _Float16* __restrict__ B1,
                    const _Float16* __restrict__ A2, size_t sA2, int k2,
                    const _Float16* __restrict__ B2,
                    const float* __restrict__ bias,
                    const float* __restrict__ lng, const float* __restrict__ lnb,
                    _Float16* __restrict__ dst, size_t sDst) {
  const int lane = threadIdx.x & 31;
  const int wave = threadIdx.x >> 5;
  const int mt   = blockIdx.x;
  const int col  = lane & 15;
  const int half = lane >> 4;

  v8f acc[4];
  for (int tt = 0; tt < 4; ++tt) {
    const int nt = wave * 4 + tt;
    v8f c = {};
    const _Float16* a1p = A1 + (size_t)mt * 512 + lane * 16;
    const _Float16* b1p = B1 + (size_t)nt * k1 * 512 + lane * 16;
#pragma unroll 4
    for (int ks = 0; ks < k1; ++ks) {
      v16h a = *(const v16h*)(a1p + (size_t)ks * sA1);
      v16h w = *(const v16h*)(b1p + (size_t)ks * 512);
      c = wmma32(a, w, c);
    }
    if (A2) {
      const _Float16* a2p = A2 + (size_t)mt * 512 + lane * 16;
      const _Float16* b2p = B2 + (size_t)nt * k2 * 512 + lane * 16;
#pragma unroll 4
      for (int ks = 0; ks < k2; ++ks) {
        v16h a = *(const v16h*)(a2p + (size_t)ks * sA2);
        v16h w = *(const v16h*)(b2p + (size_t)ks * 512);
        c = wmma32(a, w, c);
      }
    }
    const float bv = bias[nt * 16 + col];   // pre-LN bias (per feature)
#pragma unroll
    for (int r = 0; r < 8; ++r) c[r] += bv;
    acc[tt] = c;
  }

  // LayerNorm over the 1024-wide row: shuffle-reduce within wave halves, then
  // combine the 16 waves through LDS.
  __shared__ float psum[16][16];
  __shared__ float psq[16][16];
  __shared__ float smean[16];
  __shared__ float srstd[16];

  float p[8], q[8];
#pragma unroll
  for (int r = 0; r < 8; ++r) {
    p[r] = acc[0][r] + acc[1][r] + acc[2][r] + acc[3][r];
    q[r] = acc[0][r] * acc[0][r] + acc[1][r] * acc[1][r] +
           acc[2][r] * acc[2][r] + acc[3][r] * acc[3][r];
  }
#pragma unroll
  for (int m = 8; m >= 1; m >>= 1) {
#pragma unroll
    for (int r = 0; r < 8; ++r) {
      p[r] += __shfl_xor(p[r], m, 16);
      q[r] += __shfl_xor(q[r], m, 16);
    }
  }
  if (col == 0) {
#pragma unroll
    for (int r = 0; r < 8; ++r) {
      psum[wave][half * 8 + r] = p[r];
      psq [wave][half * 8 + r] = q[r];
    }
  }
  __syncthreads();
  if (threadIdx.x < 16) {
    float s = 0.f, s2 = 0.f;
    for (int w = 0; w < 16; ++w) { s += psum[w][threadIdx.x]; s2 += psq[w][threadIdx.x]; }
    float mean = s * (1.f / 1024.f);
    float var  = s2 * (1.f / 1024.f) - mean * mean;
    smean[threadIdx.x] = mean;
    srstd[threadIdx.x] = rsqrtf(var + 1e-3f);
  }
  __syncthreads();

  for (int tt = 0; tt < 4; ++tt) {
    const int cc = (wave * 4 + tt) * 16 + col;
    const float gv = lng[cc], bv = lnb[cc];
#pragma unroll
    for (int r = 0; r < 8; ++r) {
      const int rr = half * 8 + r;
      float x = (acc[tt][r] - smean[rr]) * srstd[rr] * gv + bv;
      float y = x > 0.f ? x : (expf(x) - 1.f);   // ELU
      packA_store(dst, sDst, mt, rr, cc, y);
    }
  }
}

// ---------------------------------------------------------------------------
// Fused GRU cell: one wave per 16x16 output tile, 6 accumulators
// (ir,iz,in from x@wi; hr,hz,hn from h@wh), gate math in epilogue.
// Writes fp32 h_new into the features output and packed f16 fragments into
// the next-step buffer and the all-steps buffer (for the prior head).
// ---------------------------------------------------------------------------
__global__ __launch_bounds__(256)
void gru_kernel(const _Float16* __restrict__ A1, size_t sA1,
                const _Float16* __restrict__ A2, size_t sA2,
                const _Float16* __restrict__ Wi, const _Float16* __restrict__ Wh,
                const float* __restrict__ bi, const float* __restrict__ bh,
                const float* __restrict__ hold, int holdStride,
                float* __restrict__ fout, int foutStride,
                _Float16* __restrict__ dst1, size_t sD1,
                _Float16* __restrict__ dst2, size_t sD2) {
  const int lane = threadIdx.x & 31;
  const int tile = blockIdx.x * 8 + (threadIdx.x >> 5);
  const int mt = tile >> 6;
  const int nt = tile & 63;
  const int col = lane & 15;
  const int half = lane >> 4;

  v8f cir = {}, ciz = {}, cin = {}, chr_ = {}, chz = {}, chn = {};
  const _Float16* a1p = A1 + (size_t)mt * 512 + lane * 16;
  const _Float16* a2p = A2 + (size_t)mt * 512 + lane * 16;
  const _Float16* wir = Wi + (size_t)(nt      ) * 32 * 512 + lane * 16;
  const _Float16* wiz = Wi + (size_t)(nt +  64) * 32 * 512 + lane * 16;
  const _Float16* win = Wi + (size_t)(nt + 128) * 32 * 512 + lane * 16;
  const _Float16* whr = Wh + (size_t)(nt      ) * 32 * 512 + lane * 16;
  const _Float16* whz = Wh + (size_t)(nt +  64) * 32 * 512 + lane * 16;
  const _Float16* whn = Wh + (size_t)(nt + 128) * 32 * 512 + lane * 16;

#pragma unroll 2
  for (int ks = 0; ks < 32; ++ks) {
    v16h a = *(const v16h*)(a1p + (size_t)ks * sA1);
    cir  = wmma32(a, *(const v16h*)(wir + (size_t)ks * 512), cir);
    ciz  = wmma32(a, *(const v16h*)(wiz + (size_t)ks * 512), ciz);
    cin  = wmma32(a, *(const v16h*)(win + (size_t)ks * 512), cin);
    v16h h = *(const v16h*)(a2p + (size_t)ks * sA2);
    chr_ = wmma32(h, *(const v16h*)(whr + (size_t)ks * 512), chr_);
    chz  = wmma32(h, *(const v16h*)(whz + (size_t)ks * 512), chz);
    chn  = wmma32(h, *(const v16h*)(whn + (size_t)ks * 512), chn);
  }

  const int c = nt * 16 + col;
#pragma unroll
  for (int r = 0; r < 8; ++r) {
    const int rr = half * 8 + r;
    const int m  = mt * 16 + rr;
    float gir = cir[r]  + bi[c];
    float giz = ciz[r]  + bi[c + 1024];
    float gin = cin[r]  + bi[c + 2048];
    float ghr = chr_[r] + bh[c];
    float ghz = chz[r]  + bh[c + 1024];
    float ghn = chn[r]  + bh[c + 2048];
    float rg = sigmoid_f(gir + ghr);
    float u  = sigmoid_f(giz + ghz);
    float n  = tanhf(gin + rg * ghn);
    float ho = hold[(size_t)m * holdStride + c];
    float hv = (1.f - u) * n + u * ho;
    fout[(size_t)m * foutStride + c] = hv;
    packA_store(dst1, sD1, mt, rr, c, hv);
    packA_store(dst2, sD2, mt, rr, c, hv);
  }
}

// ---------------------------------------------------------------------------
// Posterior logits GEMM + gumbel-argmax one-hot sample.
// Forward value of the straight-through sample is exactly the one-hot.
// ---------------------------------------------------------------------------
__global__ __launch_bounds__(512)
void gemm_sample_kernel(const _Float16* __restrict__ A, size_t sA,
                        const _Float16* __restrict__ B,
                        const float* __restrict__ bias,
                        const float* __restrict__ unoise,
                        float* __restrict__ posts,
                        float* __restrict__ samples,
                        float* __restrict__ featz,   // features + col 2048
                        _Float16* __restrict__ dstZ, size_t sZ) {
  const int lane = threadIdx.x & 31;
  const int wave = threadIdx.x >> 5;
  const int mt   = blockIdx.x;
  const int col  = lane & 15;
  const int half = lane >> 4;

  v8f acc[4];
  for (int tt = 0; tt < 4; ++tt) {
    const int nt = wave * 4 + tt;
    v8f cacc = {};
    const _Float16* ap = A + (size_t)mt * 512 + lane * 16;
    const _Float16* bp = B + (size_t)nt * 32 * 512 + lane * 16;
#pragma unroll 4
    for (int ks = 0; ks < 32; ++ks) {
      v16h a = *(const v16h*)(ap + (size_t)ks * sA);
      v16h w = *(const v16h*)(bp + (size_t)ks * 512);
      cacc = wmma32(a, w, cacc);
    }
    acc[tt] = cacc;
  }

  float pert[4][8];
  for (int tt = 0; tt < 4; ++tt) {
    const int cc = (wave * 4 + tt) * 16 + col;
    const float bv = bias[cc];
#pragma unroll
    for (int r = 0; r < 8; ++r) {
      const int rr = half * 8 + r;
      const int m  = mt * 16 + rr;
      float pp = acc[tt][r] + bv;
      posts[(size_t)m * 1024 + cc] = pp;
      float u = unoise[(size_t)m * 1024 + cc];
      float g = -logf(-logf(u + 1e-9f) + 1e-9f);
      pert[tt][r] = pp + g;
    }
  }

  // argmax over each 32-wide categorical group (2 tiles per group).
  for (int grp = 0; grp < 2; ++grp) {
    const int colbase = (wave * 4 + 2 * grp) * 16;
#pragma unroll
    for (int r = 0; r < 8; ++r) {
      float v0 = pert[2 * grp][r];
      float v1 = pert[2 * grp + 1][r];
      float bv; int bidx;
      if (v1 > v0) { bv = v1; bidx = 16 + col; } else { bv = v0; bidx = col; }
#pragma unroll
      for (int msk = 8; msk >= 1; msk >>= 1) {
        float ov = __shfl_xor(bv, msk, 16);
        int   oi = __shfl_xor(bidx, msk, 16);
        if (ov > bv || (ov == bv && oi < bidx)) { bv = ov; bidx = oi; }
      }
      const int rr = half * 8 + r;
      const int m  = mt * 16 + rr;
      float s0 = (bidx == col)        ? 1.f : 0.f;
      float s1 = (bidx == 16 + col)   ? 1.f : 0.f;
      const int c0 = colbase + col, c1 = c0 + 16;
      samples[(size_t)m * 1024 + c0] = s0;
      samples[(size_t)m * 1024 + c1] = s1;
      featz[(size_t)m * 3072 + c0] = s0;
      featz[(size_t)m * 3072 + c1] = s1;
      packA_store(dstZ, sZ, mt, rr, c0, s0);
      packA_store(dstZ, sZ, mt, rr, c1, s1);
    }
  }
}

// ---------------------------------------------------------------------------
// Plain GEMM + bias -> fp32 (prior head output). One wave per 16x16 tile.
// ---------------------------------------------------------------------------
__global__ __launch_bounds__(256)
void gemm_bias_kernel(const _Float16* __restrict__ A, size_t sA, int kslabs,
                      const _Float16* __restrict__ B,
                      const float* __restrict__ bias,
                      float* __restrict__ out, int ldo) {
  const int lane = threadIdx.x & 31;
  const int tile = blockIdx.x * 8 + (threadIdx.x >> 5);
  const int mt = tile >> 6;
  const int nt = tile & 63;
  const int col = lane & 15;
  const int half = lane >> 4;

  v8f cacc = {};
  const _Float16* ap = A + (size_t)mt * 512 + lane * 16;
  const _Float16* bp = B + (size_t)nt * kslabs * 512 + lane * 16;
#pragma unroll 4
  for (int ks = 0; ks < kslabs; ++ks) {
    v16h a = *(const v16h*)(ap + (size_t)ks * sA);
    v16h w = *(const v16h*)(bp + (size_t)ks * 512);
    cacc = wmma32(a, w, cacc);
  }
  const int cc = nt * 16 + col;
  const float bv = bias[cc];
#pragma unroll
  for (int r = 0; r < 8; ++r) {
    const int m = mt * 16 + half * 8 + r;
    out[(size_t)m * ldo + cc] = cacc[r] + bv;
  }
}

// ---------------------------------------------------------------------------
// Final-state copies: last h / last sample from features[T-1].
// ---------------------------------------------------------------------------
__global__ void copy_last_kernel(const float* __restrict__ feat,
                                 float* __restrict__ lasth,
                                 float* __restrict__ lasts) {
  size_t total = (size_t)64 * 3072;
  for (size_t i = (size_t)blockIdx.x * blockDim.x + threadIdx.x; i < total;
       i += (size_t)gridDim.x * blockDim.x) {
    int b = (int)(i / 3072);
    int j = (int)(i % 3072);
    float v = feat[(size_t)(63 * 64 + b) * 3072 + j];
    if (j < 2048) lasth[(size_t)b * 2048 + j] = v;
    else          lasts[(size_t)b * 1024 + (j - 2048)] = v;
  }
}

// ---------------------------------------------------------------------------
// Host launcher
// ---------------------------------------------------------------------------
extern "C" void kernel_launch(void* const* d_in, const int* in_sizes, int n_in,
                              void* d_out, int out_size, void* d_ws, size_t ws_size,
                              hipStream_t stream) {
  (void)in_sizes; (void)n_in; (void)out_size; (void)ws_size;

  const float* actions  = (const float*)d_in[0];
  /* resets (d_in[1]) are all-false and unused by the reference math */
  const float* in_h     = (const float*)d_in[2];
  const float* in_z     = (const float*)d_in[3];
  const float* embeds   = (const float*)d_in[4];
  const float* unoise   = (const float*)d_in[5];
  const float* w_z      = (const float*)d_in[6];
  const float* b_z      = (const float*)d_in[7];
  const float* w_a      = (const float*)d_in[8];
  const float* ln_in_g  = (const float*)d_in[9];
  const float* ln_in_b  = (const float*)d_in[10];
  const float* g0wi     = (const float*)d_in[11];
  const float* g0wh     = (const float*)d_in[12];
  const float* g0bi     = (const float*)d_in[13];
  const float* g0bh     = (const float*)d_in[14];
  const float* g1wi     = (const float*)d_in[15];
  const float* g1wh     = (const float*)d_in[16];
  const float* g1bi     = (const float*)d_in[17];
  const float* g1bh     = (const float*)d_in[18];
  const float* w_post_h = (const float*)d_in[19];
  const float* b_post_h = (const float*)d_in[20];
  const float* w_post_e = (const float*)d_in[21];
  const float* ln_po_g  = (const float*)d_in[22];
  const float* ln_po_b  = (const float*)d_in[23];
  const float* w_post   = (const float*)d_in[24];
  const float* b_post   = (const float*)d_in[25];
  const float* w_prior_h= (const float*)d_in[26];
  const float* b_prior_h= (const float*)d_in[27];
  const float* ln_pr_g  = (const float*)d_in[28];
  const float* ln_pr_b  = (const float*)d_in[29];
  const float* w_prior  = (const float*)d_in[30];
  const float* b_prior  = (const float*)d_in[31];

  float* out     = (float*)d_out;
  float* priors  = out;                 // [64,64,1024]
  float* posts   = out + 4194304;       // [64,64,1024]
  float* samples = out + 8388608;       // [64,64,1024]
  float* feat    = out + 12582912;      // [64,64,3072]
  float* lasth   = out + 25165824;      // [64,2048]
  float* lasts   = out + 25296896;      // [64,1024]

  _Float16* ws = (_Float16*)d_ws;
  size_t off = 0;
  auto halloc = [&](size_t halves) { _Float16* p = ws + off; off += halves; return p; };

  // packed weights
  _Float16* Wz   = halloc(1048576);   // 64nt x 32ks
  _Float16* Wa   = halloc(32768);     // 64nt x 1ks (K=16 zero-padded)
  _Float16* Wi0  = halloc(3145728);   // 192nt x 32ks
  _Float16* Wh0  = halloc(3145728);
  _Float16* Wi1  = halloc(3145728);
  _Float16* Wh1  = halloc(3145728);
  _Float16* Wph  = halloc(2097152);   // 64nt x 64ks
  _Float16* Wpe  = halloc(1572864);   // 64nt x 48ks
  _Float16* Wp   = halloc(1048576);   // 64nt x 32ks
  _Float16* Wprh = halloc(2097152);   // 64nt x 64ks
  _Float16* Wpr  = halloc(1048576);   // 64nt x 32ks
  // packed activations
  _Float16* Az     = halloc(65536);    // z one-hot, 32ks x 4mt
  _Float16* AaAll  = halloc(131072);   // actions all T, 1ks x 256mt
  _Float16* Aza    = halloc(65536);    // za, 32ks x 4mt
  _Float16* AhnA   = halloc(131072);   // hn ping, 64ks x 4mt
  _Float16* AhnB   = halloc(131072);   // hn pong
  _Float16* AeAll  = halloc(6291456);  // embeds all T, 48ks x 256mt
  _Float16* Ax     = halloc(65536);    // post-LN x, 32ks x 4mt
  _Float16* AhnAll = halloc(8388608);  // all hn, 64ks x 256mt
  _Float16* Axp    = halloc(4194304);  // prior-LN x, 32ks x 256mt

  const int PB = 256;
  auto gw = [&](size_t total) { return dim3((unsigned)((total + PB - 1) / PB)); };

  // ---- pack weights (re-done every launch: deterministic, L2-resident) ----
  pack_w_kernel<<<gw(1048576), PB, 0, stream>>>(w_z,      Wz,  32, 1024, 1024);
  pack_w_kernel<<<gw(32768),   PB, 0, stream>>>(w_a,      Wa,   1,   16, 1024);
  pack_w_kernel<<<gw(3145728), PB, 0, stream>>>(g0wi,     Wi0, 32, 1024, 3072);
  pack_w_kernel<<<gw(3145728), PB, 0, stream>>>(g0wh,     Wh0, 32, 1024, 3072);
  pack_w_kernel<<<gw(3145728), PB, 0, stream>>>(g1wi,     Wi1, 32, 1024, 3072);
  pack_w_kernel<<<gw(3145728), PB, 0, stream>>>(g1wh,     Wh1, 32, 1024, 3072);
  pack_w_kernel<<<gw(2097152), PB, 0, stream>>>(w_post_h, Wph, 64, 2048, 1024);
  pack_w_kernel<<<gw(1572864), PB, 0, stream>>>(w_post_e, Wpe, 48, 1536, 1024);
  pack_w_kernel<<<gw(1048576), PB, 0, stream>>>(w_post,   Wp,  32, 1024, 1024);
  pack_w_kernel<<<gw(2097152), PB, 0, stream>>>(w_prior_h,Wprh,64, 2048, 1024);
  pack_w_kernel<<<gw(1048576), PB, 0, stream>>>(w_prior,  Wpr, 32, 1024, 1024);

  // ---- pack initial state + whole-sequence inputs ----
  pack_a_kernel<<<gw(65536),   PB, 0, stream>>>(in_z,    1024, Az,    4,  32, 1024);
  pack_a_kernel<<<gw(131072),  PB, 0, stream>>>(in_h,    2048, AhnA,  4,  64, 2048);
  pack_a_kernel<<<gw(131072),  PB, 0, stream>>>(actions,   16, AaAll, 256, 1,   16);
  pack_a_kernel<<<gw(6291456), PB, 0, stream>>>(embeds,  1536, AeAll, 256, 48, 1536);

  const size_t S4   = 4 * 512;      // slab stride, MT=4
  const size_t S256 = 256 * 512;    // slab stride, MT=256
  _Float16* hnCur = AhnA;
  _Float16* hnNxt = AhnB;

  for (int t = 0; t < 64; ++t) {
    const float* holdBase = t ? (feat + (size_t)(t - 1) * 64 * 3072) : in_h;
    const int holdStride = t ? 3072 : 2048;
    float* featT = feat + (size_t)t * 64 * 3072;

    // za = elu(LN(z@Wz + a@Wa + b_z))
    gemm_ln_kernel<<<4, 512, 0, stream>>>(
        Az, S4, 32, Wz,
        AaAll + (size_t)t * S4, S256, 1, Wa,
        b_z, ln_in_g, ln_in_b, Aza, S4);

    // GRU0: h0' = gru(za, h0)
    gru_kernel<<<32, 256, 0, stream>>>(
        Aza, S4, hnCur, S4, Wi0, Wh0, g0bi, g0bh,
        holdBase, holdStride, featT, 3072,
        hnNxt, S4, AhnAll + (size_t)t * S4, S256);

    // GRU1: h1' = gru(h0', h1)
    gru_kernel<<<32, 256, 0, stream>>>(
        hnNxt, S4, hnCur + 32 * S4, S4, Wi1, Wh1, g1bi, g1bh,
        holdBase + 1024, holdStride, featT + 1024, 3072,
        hnNxt + 32 * S4, S4, AhnAll + 32 * S256 + (size_t)t * S4, S256);

    // x = elu(LN(hn@Wph + e@Wpe + b_post_h))
    gemm_ln_kernel<<<4, 512, 0, stream>>>(
        hnNxt, S4, 64, Wph,
        AeAll + (size_t)t * S4, S256, 48, Wpe,
        b_post_h, ln_po_g, ln_po_b, Ax, S4);

    // pp = x@Wp + b_post ; sample = onehot(argmax(pp + gumbel)) per 32-group
    gemm_sample_kernel<<<4, 512, 0, stream>>>(
        Ax, S4, Wp, b_post,
        unoise + (size_t)t * 65536,
        posts + (size_t)t * 65536,
        samples + (size_t)t * 65536,
        featT + 2048,
        Az, S4);

    _Float16* tmp = hnCur; hnCur = hnNxt; hnNxt = tmp;
  }

  // priors = elu(LN(states_h@Wprh + b_prior_h)) @ Wpr + b_prior   (all T*B rows)
  gemm_ln_kernel<<<256, 512, 0, stream>>>(
      AhnAll, S256, 64, Wprh,
      nullptr, 0, 0, nullptr,
      b_prior_h, ln_pr_g, ln_pr_b, Axp, S256);
  gemm_bias_kernel<<<2048, 256, 0, stream>>>(
      Axp, S256, 32, Wpr, b_prior, priors, 1024);

  copy_last_kernel<<<gw(64 * 3072), PB, 0, stream>>>(feat, lasth, lasts);
}
